// Decoder_65025804861623
// MI455X (gfx1250) — matrix operand order repaired and checked by reading, not compiled, not run on hardware
//
#include <hip/hip_runtime.h>

// ---------------------------------------------------------------------------
// Decoder (GRU + Luong attention + greedy feedback), 129 sequential steps.
// Strategy: bf16 WMMA (v_wmma_f32_16x16x32_bf16) for all GEMMs, fp32 VALU for
// gates/softmax/attention. Weights converted fp32->bf16 once per launch into
// d_ws (all weights are L2-resident: 192MB L2 >> ~37MB working set).
// ---------------------------------------------------------------------------

typedef __attribute__((ext_vector_type(16))) __bf16       v16bf;
typedef __attribute__((ext_vector_type(8)))  float        v8f;
typedef __attribute__((ext_vector_type(4)))  unsigned int v4u;
typedef unsigned short u16;

#define V_SZ   32000
#define E_SZ   256
#define H_SZ   512
#define T_ENC  128
#define T_DEC  129
#define B_SZ   64
#define SOS    2

__device__ __forceinline__ u16 f2bf(float f) {
  unsigned int u = __float_as_uint(f);
  // round-to-nearest-even fp32 -> bf16
  return (u16)((u + 0x7FFFu + ((u >> 16) & 1u)) >> 16);
}

union Frag { v4u q[2]; v16bf v; };

// ---------------------------------------------------------------------------
// fp32 -> bf16 conversion (weight prep, once per launch)
// ---------------------------------------------------------------------------
__global__ void __launch_bounds__(256)
k_cvt(const float* __restrict__ src, u16* __restrict__ dst, int n) {
  int i = blockIdx.x * 256 + threadIdx.x;
  if (i < n) dst[i] = f2bf(src[i]);
}

// ---------------------------------------------------------------------------
// init: h = encoder_hidden[0]; concat[:, :512] = bf16(h); x = bf16(emb[SOS])
// ---------------------------------------------------------------------------
__global__ void __launch_bounds__(256)
k_init(const float* __restrict__ enc_hidden, const float* __restrict__ emb,
       float* __restrict__ h, u16* __restrict__ concat, u16* __restrict__ x) {
  int idx = blockIdx.x * 256 + threadIdx.x;
  if (idx < B_SZ * H_SZ) {
    float v = enc_hidden[idx];          // layer 0 == first B*H elements
    h[idx] = v;
    int b = idx >> 9, j = idx & 511;
    concat[b * 1024 + j] = f2bf(v);
  } else if (idx < B_SZ * H_SZ + B_SZ * E_SZ) {
    int i = idx - B_SZ * H_SZ;
    x[i] = f2bf(emb[SOS * E_SZ + (i & (E_SZ - 1))]);
  }
}

// ---------------------------------------------------------------------------
// Generic WMMA GEMM: out[64 x N] = A[64 x K](bf16) @ W[N x K](bf16, row-major)^T
// Block = 256 threads = 8 waves. Each block owns 64 N-columns (4 N-tiles);
// waves 0-3: M-tiles {0,1}, waves 4-7: M-tiles {2,3}. K%32 == 0.
// A-frag (16-bit A 16x32): lane<16 -> K {kk..kk+7, kk+16..kk+23} of row M=lane
//                          lane>=16 -> K {kk+8..kk+15, kk+24..kk+31}
// B-frag (16-bit B 32x16): lane<16 -> K {kk..kk+7, kk+16..kk+23} of col N=lane (== row of W)
//                          lane>=16 -> K {kk+8..kk+15, kk+24..kk+31}
// D (8xVGPR f32): lane<16 -> row i, col lane; lane>=16 -> row 8+i, col lane-16
// ---------------------------------------------------------------------------
__global__ void __launch_bounds__(256)
k_gemm(const u16* __restrict__ A, int lda,
       const u16* __restrict__ W, int ldw, int K,
       float* __restrict__ outF, long ostrideF,
       u16* __restrict__ outB, int ldoB, int act) {
  const int wv    = threadIdx.x >> 5;
  const int lane  = threadIdx.x & 31;
  const int lhalf = lane >> 4;
  const int l15   = lane & 15;
  const int nt    = blockIdx.x * 4 + (wv & 3);
  const int mt0   = (wv >> 2) * 2;
  const int col   = nt * 16 + l15;

  v8f c0 = {0.f, 0.f, 0.f, 0.f, 0.f, 0.f, 0.f, 0.f};
  v8f c1 = c0;

  const u16* wrow  = W + (size_t)col * ldw + lhalf * 8;
  const u16* arow0 = A + (size_t)(mt0 * 16 + l15) * lda + lhalf * 8;
  const u16* arow1 = arow0 + (size_t)16 * lda;

  for (int kk = 0; kk < K; kk += 32) {
    Frag bf, a0, a1;
    bf.q[0] = *(const v4u*)(wrow + kk);
    bf.q[1] = *(const v4u*)(wrow + kk + 16);
    a0.q[0] = *(const v4u*)(arow0 + kk);
    a0.q[1] = *(const v4u*)(arow0 + kk + 16);
    a1.q[0] = *(const v4u*)(arow1 + kk);
    a1.q[1] = *(const v4u*)(arow1 + kk + 16);
    c0 = __builtin_amdgcn_wmma_f32_16x16x32_bf16(false, a0.v, false, bf.v,
                                                 (short)0, c0, false, false);
    c1 = __builtin_amdgcn_wmma_f32_16x16x32_bf16(false, a1.v, false, bf.v,
                                                 (short)0, c1, false, false);
  }

  const int rbase = mt0 * 16 + lhalf * 8;
#pragma unroll
  for (int i = 0; i < 8; ++i) {
    float v0 = c0[i], v1 = c1[i];
    if (act) { v0 = tanhf(v0); v1 = tanhf(v1); }
    int r0 = rbase + i, r1 = r0 + 16;
    if (outF) {
      outF[(size_t)r0 * ostrideF + col] = v0;
      outF[(size_t)r1 * ostrideF + col] = v1;
    }
    if (outB) {
      outB[(size_t)r0 * ldoB + col] = f2bf(v0);
      outB[(size_t)r1 * ldoB + col] = f2bf(v1);
    }
  }
}

// ---------------------------------------------------------------------------
// Per-batch-row: GRU gates (reads gi/gh from WMMA GEMMs) -> h_new, then Luong
// attention (q matvec, scores, softmax, ctx) -> concat = [h_new | ctx] (bf16).
// 64 blocks (one per b) x 128 threads.
// ---------------------------------------------------------------------------
__global__ void __launch_bounds__(128)
k_gru_attn(const float* __restrict__ gi, const float* __restrict__ gh,
           const float* __restrict__ b_ih, const float* __restrict__ b_hh,
           float* __restrict__ h, u16* __restrict__ concat,
           const float* __restrict__ wa_attn, const float* __restrict__ enc) {
  const int b = blockIdx.x, tid = threadIdx.x;
  __shared__ float sh[H_SZ];     // h_new
  __shared__ float sq[H_SZ];     // q
  __shared__ float ssc[T_ENC];   // scores / attn
  __shared__ float sred[T_ENC];  // reduction scratch

  // GRU elementwise (PyTorch gate order r,z,n)
  for (int j = tid; j < H_SZ; j += 128) {
    float ir = gi[b * 1536 + j] + b_ih[j];
    float iz = gi[b * 1536 + 512 + j] + b_ih[512 + j];
    float in = gi[b * 1536 + 1024 + j] + b_ih[1024 + j];
    float hr = gh[b * 1536 + j] + b_hh[j];
    float hz = gh[b * 1536 + 512 + j] + b_hh[512 + j];
    float hn = gh[b * 1536 + 1024 + j] + b_hh[1024 + j];
    float r = 1.f / (1.f + expf(-(ir + hr)));
    float z = 1.f / (1.f + expf(-(iz + hz)));
    float n = tanhf(in + r * hn);
    float hv = (1.f - z) * n + z * h[b * H_SZ + j];
    sh[j] = hv;
  }
  __syncthreads();
  for (int j = tid; j < H_SZ; j += 128) {
    h[b * H_SZ + j] = sh[j];
    concat[b * 1024 + j] = f2bf(sh[j]);
  }
  // q[e] = sum_k h_new[k] * wa_attn[k, e]   (coalesced across threads per k)
  for (int e = tid; e < H_SZ; e += 128) {
    float acc = 0.f;
    for (int k = 0; k < H_SZ; ++k) acc += sh[k] * wa_attn[k * H_SZ + e];
    sq[e] = acc;
  }
  __syncthreads();
  // scores[t] = dot(q, enc[b, t, :])
  {
    float acc = 0.f;
    const float* ep = enc + ((size_t)b * T_ENC + tid) * H_SZ;
    for (int k = 0; k < H_SZ; ++k) acc += sq[k] * ep[k];
    ssc[tid] = acc;
  }
  sred[tid] = ssc[tid];
  __syncthreads();
  for (int s = 64; s > 0; s >>= 1) {
    if (tid < s) sred[tid] = fmaxf(sred[tid], sred[tid + s]);
    __syncthreads();
  }
  float mx = sred[0];
  __syncthreads();
  float ev = expf(ssc[tid] - mx);
  sred[tid] = ev;
  __syncthreads();
  for (int s = 64; s > 0; s >>= 1) {
    if (tid < s) sred[tid] += sred[tid + s];
    __syncthreads();
  }
  float inv = 1.f / sred[0];
  __syncthreads();
  ssc[tid] = ev * inv;
  __syncthreads();
  // ctx[e] = sum_t attn[t] * enc[b, t, e]  -> concat upper half (bf16)
  for (int e = tid; e < H_SZ; e += 128) {
    float acc = 0.f;
    const float* ep = enc + (size_t)b * T_ENC * H_SZ + e;
    for (int t2 = 0; t2 < T_ENC; ++t2) acc += ssc[t2] * ep[t2 * H_SZ];
    concat[b * 1024 + H_SZ + e] = f2bf(acc);
  }
}

// ---------------------------------------------------------------------------
// Per-row (64 blocks x 256 thr): max+argmax over V, log-sum-exp, in-place
// log-softmax fixup of d_out row, and gather next-step embedding (feedback).
// ---------------------------------------------------------------------------
__global__ void __launch_bounds__(256)
k_logsoftmax_argmax(float* __restrict__ row_base, long brow_stride,
                    const float* __restrict__ emb, u16* __restrict__ x_bf16) {
  const int b = blockIdx.x, tid = threadIdx.x;
  float* row = row_base + (size_t)b * brow_stride;
  __shared__ float smax[256];
  __shared__ int   sidx[256];
  __shared__ float ssum[256];

  float lm = -3.402823466e38f; int li = 0;
  for (int c = tid; c < V_SZ; c += 256) {
    float v = row[c];
    if (v > lm) { lm = v; li = c; }
  }
  smax[tid] = lm; sidx[tid] = li;
  __syncthreads();
  for (int s = 128; s > 0; s >>= 1) {
    if (tid < s) {
      float a = smax[tid], bv = smax[tid + s];
      int ai = sidx[tid], bi = sidx[tid + s];
      if (bv > a || (bv == a && bi < ai)) { smax[tid] = bv; sidx[tid] = bi; }
    }
    __syncthreads();
  }
  const float rmax = smax[0];
  const int   ridx = sidx[0];
  __syncthreads();

  float ls = 0.f;
  for (int c = tid; c < V_SZ; c += 256) ls += expf(row[c] - rmax);
  ssum[tid] = ls;
  __syncthreads();
  for (int s = 128; s > 0; s >>= 1) {
    if (tid < s) ssum[tid] += ssum[tid + s];
    __syncthreads();
  }
  const float sub = rmax + logf(ssum[0]);

  for (int c = tid; c < V_SZ; c += 256) row[c] -= sub;   // logp in place

  // greedy feedback: x_{t+1} = bf16(emb[argmax])
  if (tid < E_SZ) x_bf16[b * E_SZ + tid] = f2bf(emb[(size_t)ridx * E_SZ + tid]);
}

__global__ void __launch_bounds__(256)
k_hid_out(const float* __restrict__ h, float* __restrict__ dst) {
  int i = blockIdx.x * 256 + threadIdx.x;
  if (i < B_SZ * H_SZ) dst[i] = h[i];
}

// ---------------------------------------------------------------------------
// host launcher
// ---------------------------------------------------------------------------
extern "C" void kernel_launch(void* const* d_in, const int* in_sizes, int n_in,
                              void* d_out, int out_size, void* d_ws, size_t ws_size,
                              hipStream_t stream) {
  const float* enc_hidden = (const float*)d_in[0];
  const float* enc_out    = (const float*)d_in[1];
  // d_in[2] = target: unused by the reference computation (SOS start, greedy feedback)
  const float* emb     = (const float*)d_in[3];
  const float* w_ih    = (const float*)d_in[4];
  const float* w_hh    = (const float*)d_in[5];
  const float* b_ih    = (const float*)d_in[6];
  const float* b_hh    = (const float*)d_in[7];
  const float* wa_attn = (const float*)d_in[8];
  const float* wa      = (const float*)d_in[9];
  const float* fc      = (const float*)d_in[10];
  float* out = (float*)d_out;

  // workspace carve-up (~37 MB; L2-resident on MI455X)
  char* ws = (char*)d_ws;
  size_t off = 0;
  auto take = [&](size_t bytes) -> void* {
    void* p = ws + off;
    off = (off + bytes + 255) & ~(size_t)255;
    return p;
  };
  u16*   fc_b     = (u16*)take((size_t)V_SZ * H_SZ * 2);       // 32.8 MB
  u16*   wih_b    = (u16*)take((size_t)3 * H_SZ * E_SZ * 2);
  u16*   whh_b    = (u16*)take((size_t)3 * H_SZ * H_SZ * 2);
  u16*   wa_b     = (u16*)take((size_t)H_SZ * 1024 * 2);
  u16*   x_b      = (u16*)take((size_t)B_SZ * E_SZ * 2);
  u16*   concat_b = (u16*)take((size_t)B_SZ * 1024 * 2);
  u16*   o_b      = (u16*)take((size_t)B_SZ * H_SZ * 2);
  float* h_f      = (float*)take((size_t)B_SZ * H_SZ * 4);
  float* gi_f     = (float*)take((size_t)B_SZ * 1536 * 4);
  float* gh_f     = (float*)take((size_t)B_SZ * 1536 * 4);

  // ---- prep: weight conversion + state init (deterministic, every call) ----
  {
    int n;
    n = V_SZ * H_SZ;       k_cvt<<<(n + 255) / 256, 256, 0, stream>>>(fc,   fc_b,  n);
    n = 3 * H_SZ * E_SZ;   k_cvt<<<(n + 255) / 256, 256, 0, stream>>>(w_ih, wih_b, n);
    n = 3 * H_SZ * H_SZ;   k_cvt<<<(n + 255) / 256, 256, 0, stream>>>(w_hh, whh_b, n);
    n = H_SZ * 1024;       k_cvt<<<(n + 255) / 256, 256, 0, stream>>>(wa,   wa_b,  n);
    n = B_SZ * H_SZ + B_SZ * E_SZ;
    k_init<<<(n + 255) / 256, 256, 0, stream>>>(enc_hidden, emb, h_f, concat_b, x_b);
  }

  const long out_bstride = (long)T_DEC * V_SZ;  // row stride between batches in d_out

  for (int t = 0; t < T_DEC; ++t) {
    // gi = x @ w_ih^T        [64 x 1536], K=256
    k_gemm<<<1536 / 64, 256, 0, stream>>>(x_b, E_SZ, wih_b, E_SZ, E_SZ,
                                          gi_f, 1536L, (u16*)nullptr, 0, 0);
    // gh = h @ w_hh^T        [64 x 1536], K=512 (A = concat[:, :512] = bf16(h))
    k_gemm<<<1536 / 64, 256, 0, stream>>>(concat_b, 1024, whh_b, H_SZ, H_SZ,
                                          gh_f, 1536L, (u16*)nullptr, 0, 0);
    // GRU gates + attention -> h, concat = [h_new | ctx]
    k_gru_attn<<<B_SZ, 128, 0, stream>>>(gi_f, gh_f, b_ih, b_hh, h_f, concat_b,
                                         wa_attn, enc_out);
    // o = tanh(concat @ wa^T)  [64 x 512], K=1024, bf16 out
    k_gemm<<<512 / 64, 256, 0, stream>>>(concat_b, 1024, wa_b, 1024, 1024,
                                         (float*)nullptr, 0L, o_b, H_SZ, 1);
    // logits = o @ fc^T  [64 x 32000], K=512 -> raw into d_out[:, t, :]
    k_gemm<<<V_SZ / 64, 256, 0, stream>>>(o_b, H_SZ, fc_b, H_SZ, H_SZ,
                                          out + (size_t)t * V_SZ, out_bstride,
                                          (u16*)nullptr, 0, 0);
    // log-softmax (in place) + argmax -> next embedding
    k_logsoftmax_argmax<<<B_SZ, 256, 0, stream>>>(out + (size_t)t * V_SZ,
                                                  out_bstride, emb, x_b);
  }

  // decoder_hidden tail of d_out
  k_hid_out<<<(B_SZ * H_SZ + 255) / 256, 256, 0, stream>>>(
      h_f, out + (size_t)B_SZ * T_DEC * V_SZ);
}
